// GNN_34978213659188
// MI455X (gfx1250) — matrix operand order repaired and checked by reading, not compiled, not run on hardware
//
#include <hip/hip_runtime.h>
#include <hip/hip_bf16.h>

#define N_NODES 200000
#define N_EDGES 400000
#define EMB     300
#define KP1     320        // padded K for GEMM1 (agg cols)
#define NP1     640        // padded GEMM1 out cols / GEMM2 K
#define NP2     320        // padded GEMM2 out cols
#define LAYERS  5

typedef __attribute__((ext_vector_type(16))) __bf16 v16bf;
typedef __attribute__((ext_vector_type(8)))  __bf16 v8bf;
typedef __attribute__((ext_vector_type(8)))  float  v8f;

union BF16x16 { v16bf v; __bf16 e[16]; };
union F32x8   { v8f   v; float  e[8];  };

// ---------------- weight prep: transpose + bf16, zero-padded ----------------
__global__ __launch_bounds__(256)
void prep_w1_kernel(const float* __restrict__ W1, __bf16* __restrict__ Wt1) {
    int idx = blockIdx.x * 256 + threadIdx.x;            // L*NP1*KP1 = 1,024,000
    const int l = idx / (NP1 * KP1);
    const int r = idx % (NP1 * KP1);
    const int n = r / KP1;
    const int k = r % KP1;
    float v = (n < 600 && k < 300) ? W1[(size_t)l * 300 * 600 + (size_t)k * 600 + n] : 0.f;
    Wt1[idx] = (__bf16)v;
}

__global__ __launch_bounds__(256)
void prep_w2_kernel(const float* __restrict__ W2, __bf16* __restrict__ Wt2) {
    int idx = blockIdx.x * 256 + threadIdx.x;            // L*NP2*NP1 = 1,024,000
    const int l = idx / (NP2 * NP1);
    const int r = idx % (NP2 * NP1);
    const int n = r / NP1;
    const int k = r % NP1;
    float v = (n < 300 && k < 600) ? W2[(size_t)l * 600 * 300 + (size_t)k * 300 + n] : 0.f;
    Wt2[idx] = (__bf16)v;
}

// ---------------- node embedding: h = sum of 7 table lookups ----------------
__global__ __launch_bounds__(320)
void node_embed_kernel(const int* __restrict__ x,
                       const float* __restrict__ xe1, const float* __restrict__ xe2,
                       const float* __restrict__ xe3, const float* __restrict__ xe4,
                       const float* __restrict__ xe5, const float* __restrict__ xe6,
                       const float* __restrict__ xe7, float* __restrict__ h) {
    const int n = blockIdx.x;
    const int c = threadIdx.x;
    if (c >= EMB) return;
    const int* xr = x + n * 7;
    float v = xe1[(size_t)xr[0] * EMB + c] + xe2[(size_t)xr[1] * EMB + c]
            + xe3[(size_t)xr[2] * EMB + c] + xe4[(size_t)xr[3] * EMB + c]
            + xe5[(size_t)xr[4] * EMB + c] + xe6[(size_t)xr[5] * EMB + c]
            + xe7[(size_t)xr[6] * EMB + c];
    h[(size_t)n * EMB + c] = v;
}

// ---------------- zeroing ----------------
__global__ __launch_bounds__(256)
void zero_f4_kernel(float4* __restrict__ p) {        // exactly N_NODES*KP1/4 float4s
    size_t i = (size_t)blockIdx.x * 256 + threadIdx.x;
    p[i] = make_float4(0.f, 0.f, 0.f, 0.f);
}

__global__ __launch_bounds__(256)
void zero_stats_kernel(float* __restrict__ p) {      // 600 floats (colsum || colsq)
    int i = blockIdx.x * 256 + threadIdx.x;
    if (i < 600) p[i] = 0.f;
}

// ---------------- scatter: agg[dst] += h[src] + edge_embed ----------------
__global__ __launch_bounds__(320)
void scatter_kernel(const float* __restrict__ h, const int* __restrict__ ei,
                    const int* __restrict__ eattr,
                    const float* __restrict__ t1, const float* __restrict__ t2,
                    const float* __restrict__ t3, const float* __restrict__ t4,
                    float* __restrict__ agg) {
    const int e = blockIdx.x;
    int src, dst, a0, a1, a2, a3;
    if (e < N_EDGES) {
        src = ei[e]; dst = ei[N_EDGES + e];
        const int* ar = eattr + (size_t)e * 4;
        a0 = ar[0]; a1 = ar[1]; a2 = ar[2]; a3 = ar[3];
    } else {                               // self loop, attr = [4,0,0,0]
        src = dst = e - N_EDGES; a0 = 4; a1 = 0; a2 = 0; a3 = 0;
    }
    const float* r1 = t1 + (size_t)a0 * EMB;
    const float* r2 = t2 + (size_t)a1 * EMB;
    const float* r3 = t3 + (size_t)a2 * EMB;
    const float* r4 = t4 + (size_t)a3 * EMB;
    const float* hs = h + (size_t)src * EMB;
    float* ad = agg + (size_t)dst * KP1;
    const int c = threadIdx.x;
    if (c < EMB) {
        float v = hs[c] + r1[c] + r2[c] + r3[c] + r4[c];
        atomicAdd(ad + c, v);
    }
}

// ---------------- GEMM1: hid = relu(agg @ W1 + b1), bf16 out ----------------
// A: f32 [N, KP1] (cvt->bf16), B: bf16 Wt1 [NP1, KP1] (n-major), 4 col-tiles/wave
__global__ __launch_bounds__(128)
void gemm1_kernel(const float* __restrict__ A, const __bf16* __restrict__ Wt,
                  const float* __restrict__ bias, __bf16* __restrict__ H) {
    const int lane = threadIdx.x & 31;
    const int wave = threadIdx.x >> 5;                  // 0..3
    const int m    = lane & 15;
    const int hi   = lane >> 4;                         // 0 or 1
    const int rowBase = (blockIdx.x * 4 + wave) * 16;   // 12500 tiles exact
    const int colBase = blockIdx.y * 64;
    const float* Arow = A + (size_t)(rowBase + m) * KP1;

    F32x8 acc[4];
#pragma unroll
    for (int t = 0; t < 4; ++t)
#pragma unroll
        for (int i = 0; i < 8; ++i) acc[t].e[i] = 0.f;

    for (int kb = 0; kb < KP1; kb += 32) {
        // A fragment: lanes0-15 K{kb+0..15}, lanes16-31 K{kb+16..31} (contiguous)
        BF16x16 a;
        const float* p0 = Arow + kb + hi * 16;
        const float* p1 = Arow + kb + hi * 16 + 8;
#pragma unroll
        for (int i = 0; i < 8; ++i) { a.e[i] = (__bf16)p0[i]; a.e[8 + i] = (__bf16)p1[i]; }
#pragma unroll
        for (int t = 0; t < 4; ++t) {
            // B fragment: lane n=colBase+t*16+m, 16 contiguous K starting kb+hi*16
            const v16bf b = *(const v16bf*)(Wt + (size_t)(colBase + t * 16 + m) * KP1 + kb + hi * 16);
            acc[t].v = __builtin_amdgcn_wmma_f32_16x16x32_bf16(
                false, a.v, false, b, (short)0, acc[t].v, false, false);
        }
    }
#pragma unroll
    for (int t = 0; t < 4; ++t) {
        const int nc = colBase + t * 16 + m;
        const float bv = (nc < 600) ? bias[nc] : 0.f;
#pragma unroll
        for (int v = 0; v < 8; ++v) {
            const int r = rowBase + v + 8 * hi;
            float xv = acc[t].e[v] + bv;
            xv = fmaxf(xv, 0.f);
            H[(size_t)r * NP1 + nc] = (__bf16)xv;
        }
    }
}

// ---------------- GEMM2: out = hid @ W2 + b2, f32 out ----------------
__global__ __launch_bounds__(128)
void gemm2_kernel(const __bf16* __restrict__ A, const __bf16* __restrict__ Wt,
                  const float* __restrict__ bias, float* __restrict__ Out) {
    const int lane = threadIdx.x & 31;
    const int wave = threadIdx.x >> 5;
    const int m    = lane & 15;
    const int hi   = lane >> 4;
    const int rowBase = (blockIdx.x * 4 + wave) * 16;
    const int colBase = blockIdx.y * 64;
    const __bf16* Arow = A + (size_t)(rowBase + m) * NP1;

    F32x8 acc[4];
#pragma unroll
    for (int t = 0; t < 4; ++t)
#pragma unroll
        for (int i = 0; i < 8; ++i) acc[t].e[i] = 0.f;

    for (int kb = 0; kb < NP1; kb += 32) {
        BF16x16 a;
        *(v8bf*)&a.e[0] = *(const v8bf*)(Arow + kb + hi * 16);
        *(v8bf*)&a.e[8] = *(const v8bf*)(Arow + kb + hi * 16 + 8);
#pragma unroll
        for (int t = 0; t < 4; ++t) {
            const v16bf b = *(const v16bf*)(Wt + (size_t)(colBase + t * 16 + m) * NP1 + kb + hi * 16);
            acc[t].v = __builtin_amdgcn_wmma_f32_16x16x32_bf16(
                false, a.v, false, b, (short)0, acc[t].v, false, false);
        }
    }
#pragma unroll
    for (int t = 0; t < 4; ++t) {
        const int nc = colBase + t * 16 + m;
        const float bv = (nc < 300) ? bias[nc] : 0.f;
#pragma unroll
        for (int v = 0; v < 8; ++v) {
            const int r = rowBase + v + 8 * hi;
            Out[(size_t)r * NP2 + nc] = acc[t].e[v] + bv;
        }
    }
}

// ---------------- BatchNorm: column sums / sumsq then apply ----------------
__global__ __launch_bounds__(256)
void bn_stats_kernel(const float* __restrict__ Out, float* __restrict__ colsum,
                     float* __restrict__ colsq) {
    const int t  = threadIdx.x;          // cols t and t+256
    const int c1 = t + 256;
    const int r0 = blockIdx.x * 256;
    float s0 = 0.f, q0 = 0.f, s1 = 0.f, q1 = 0.f;
    for (int i = 0; i < 256; ++i) {
        const int r = r0 + i;
        if (r >= N_NODES) break;
        const float* row = Out + (size_t)r * NP2;
        const float x0 = row[t]; s0 += x0; q0 += x0 * x0;
        if (c1 < EMB) { const float x1 = row[c1]; s1 += x1; q1 += x1 * x1; }
    }
    atomicAdd(&colsum[t], s0);
    atomicAdd(&colsq[t], q0);
    if (c1 < EMB) { atomicAdd(&colsum[c1], s1); atomicAdd(&colsq[c1], q1); }
}

__global__ __launch_bounds__(320)
void bn_apply_kernel(const float* __restrict__ Out, const float* __restrict__ colsum,
                     const float* __restrict__ colsq, const float* __restrict__ gamma,
                     const float* __restrict__ beta, float* __restrict__ Hout, int do_relu) {
    const int n = blockIdx.x;
    const int c = threadIdx.x;
    if (c >= EMB) return;
    const float inv = 1.f / (float)N_NODES;
    const float mu  = colsum[c] * inv;
    const float var = colsq[c] * inv - mu * mu;
    const float rs  = rsqrtf(var + 1e-5f);
    float y = (Out[(size_t)n * NP2 + c] - mu) * rs * gamma[c] + beta[c];
    if (do_relu) y = fmaxf(y, 0.f);
    Hout[(size_t)n * EMB + c] = y;
}

// ---------------- driver ----------------
extern "C" void kernel_launch(void* const* d_in, const int* in_sizes, int n_in,
                              void* d_out, int out_size, void* d_ws, size_t ws_size,
                              hipStream_t stream) {
    const int*   x     = (const int*)d_in[0];
    const int*   ei    = (const int*)d_in[1];
    const int*   eattr = (const int*)d_in[2];
    const float* xe1 = (const float*)d_in[3];
    const float* xe2 = (const float*)d_in[4];
    const float* xe3 = (const float*)d_in[5];
    const float* xe4 = (const float*)d_in[6];
    const float* xe5 = (const float*)d_in[7];
    const float* xe6 = (const float*)d_in[8];
    const float* xe7 = (const float*)d_in[9];
    const float* e1 = (const float*)d_in[10];   // [L,6,EMB]
    const float* e2 = (const float*)d_in[11];   // [L,3,EMB]
    const float* e3 = (const float*)d_in[12];   // [L,2,EMB]
    const float* e4 = (const float*)d_in[13];   // [L,2,EMB]
    const float* W1 = (const float*)d_in[14];   // [L,300,600]
    const float* b1 = (const float*)d_in[15];   // [L,600]
    const float* W2 = (const float*)d_in[16];   // [L,600,300]
    const float* b2 = (const float*)d_in[17];   // [L,300]
    const float* gamma = (const float*)d_in[18];
    const float* beta  = (const float*)d_in[19];

    // workspace layout (all chunks 256B-aligned by construction)
    char* ws = (char*)d_ws;
    float*  h    = (float*)ws;   ws += (size_t)N_NODES * EMB * sizeof(float);   // 240 MB
    float*  agg  = (float*)ws;   ws += (size_t)N_NODES * KP1 * sizeof(float);   // 256 MB (also GEMM2 out)
    __bf16* hid  = (__bf16*)ws;  ws += (size_t)N_NODES * NP1 * sizeof(__bf16);  // 256 MB
    __bf16* Wt1  = (__bf16*)ws;  ws += (size_t)LAYERS * NP1 * KP1 * sizeof(__bf16);
    __bf16* Wt2  = (__bf16*)ws;  ws += (size_t)LAYERS * NP2 * NP1 * sizeof(__bf16);
    float*  colsum = (float*)ws; ws += 512 * sizeof(float);   // colsum[300] then colsq[300] inside 512+
    float*  colsq  = colsum + 300;
    float*  outbuf = agg;        // GEMM2 output aliases agg (dead by then)

    // once per call: bf16 transposed, zero-padded weights
    prep_w1_kernel<<<(LAYERS * NP1 * KP1) / 256, 256, 0, stream>>>(W1, Wt1);
    prep_w2_kernel<<<(LAYERS * NP2 * NP1) / 256, 256, 0, stream>>>(W2, Wt2);

    // initial node embeddings
    node_embed_kernel<<<N_NODES, 320, 0, stream>>>(x, xe1, xe2, xe3, xe4, xe5, xe6, xe7, h);

    const size_t n4 = (size_t)N_NODES * KP1 / 4;        // 16,000,000 float4s
    for (int l = 0; l < LAYERS; ++l) {
        zero_f4_kernel<<<(unsigned)(n4 / 256), 256, 0, stream>>>((float4*)agg);
        zero_stats_kernel<<<3, 256, 0, stream>>>(colsum);

        scatter_kernel<<<N_EDGES + N_NODES, 320, 0, stream>>>(
            h, ei, eattr,
            e1 + (size_t)l * 6 * EMB, e2 + (size_t)l * 3 * EMB,
            e3 + (size_t)l * 2 * EMB, e4 + (size_t)l * 2 * EMB, agg);

        gemm1_kernel<<<dim3(N_NODES / 64, NP1 / 64), 128, 0, stream>>>(
            agg, Wt1 + (size_t)l * NP1 * KP1, b1 + (size_t)l * 600, hid);

        gemm2_kernel<<<dim3(N_NODES / 64, NP2 / 64), 128, 0, stream>>>(
            hid, Wt2 + (size_t)l * NP2 * NP1, b2 + (size_t)l * 300, outbuf);

        bn_stats_kernel<<<(N_NODES + 255) / 256, 256, 0, stream>>>(outbuf, colsum, colsq);

        float* dst = (l == LAYERS - 1) ? (float*)d_out : h;
        bn_apply_kernel<<<N_NODES, 320, 0, stream>>>(
            outbuf, colsum, colsq, gamma + (size_t)l * 300, beta + (size_t)l * 300,
            dst, (l < LAYERS - 1) ? 1 : 0);
    }
}
